// AVWDCRNN_30073361006782
// MI455X (gfx1250) — matrix operand order repaired and checked by reading, not compiled, not run on hardware
//
#include <hip/hip_runtime.h>

// ---------------------------------------------------------------------------
// AGCRN forward for MI455X (gfx1250): all heavy math on v_wmma_f32_16x16x32_bf16
// N=2048 nodes, B=16 batch, T=24 steps, EMBED=16, K=3 supports, O=64
// ---------------------------------------------------------------------------

#define NN 2048      // nodes
#define BB 16        // batch
#define TT 24        // timesteps
#define OO64 64      // DIM_OUT

typedef __attribute__((ext_vector_type(16))) __bf16 v16bf;
typedef __attribute__((ext_vector_type(8)))  __bf16 v8bf;
typedef __attribute__((ext_vector_type(8)))  float  v8f;

__device__ __forceinline__ __bf16 f2bf(float f) {
    unsigned u = __builtin_bit_cast(unsigned, f);
    unsigned r = u + 0x7fffu + ((u >> 16) & 1u);       // round-to-nearest-even
    unsigned short hs = (unsigned short)(r >> 16);
    return __builtin_bit_cast(__bf16, hs);
}
__device__ __forceinline__ float bf2f(__bf16 b) {
    unsigned u = ((unsigned)__builtin_bit_cast(unsigned short, b)) << 16;
    return __builtin_bit_cast(float, u);
}

// CDNA5 16-bit A fragment (16x32): lane L = row L%16, K-runs {g*8.. , 16+g*8..}, g=L/16
__device__ __forceinline__ v16bf load_a_frag(const __bf16* rowp, int g) {
    union { v16bf v; v8bf h[2]; } u;
    u.h[0] = *(const v8bf*)(rowp + g * 8);
    u.h[1] = *(const v8bf*)(rowp + 16 + g * 8);
    return u.v;
}
// CDNA5 16-bit B fragment (32x16): lane L = col L%16, K-runs {g*8.. , 16+g*8..}, g=L/16
__device__ __forceinline__ v16bf load_b_frag(const __bf16* colp, int g) {
    union { v16bf v; v8bf h[2]; } u;
    u.h[0] = *(const v8bf*)(colp + g * 8);
    u.h[1] = *(const v8bf*)(colp + 16 + g * 8);
    return u.v;
}
__device__ __forceinline__ v8f wmma_bf16(v16bf a, v16bf b, v8f c) {
    return __builtin_amdgcn_wmma_f32_16x16x32_bf16(false, a, false, b,
                                                   (short)0, c, false, false);
}

// ---------------------------------------------------------------------------
// A = softmax(relu(E E^T)) row-wise -> bf16.  One block per row.
// ---------------------------------------------------------------------------
__global__ __launch_bounds__(256) void adj_kernel(const float* __restrict__ E,
                                                  __bf16* __restrict__ Abf) {
    __shared__ float row[NN];
    __shared__ float red[256];
    const int n = blockIdx.x, tid = threadIdx.x;
    float e[16];
#pragma unroll
    for (int d = 0; d < 16; d++) e[d] = E[n * 16 + d];
    float mx = -3.4e38f;
    for (int m = tid; m < NN; m += 256) {
        float s = 0.f;
#pragma unroll
        for (int d = 0; d < 16; d++) s += e[d] * E[m * 16 + d];
        s = fmaxf(s, 0.f);
        row[m] = s;
        mx = fmaxf(mx, s);
    }
    red[tid] = mx; __syncthreads();
    for (int o = 128; o > 0; o >>= 1) {
        if (tid < o) red[tid] = fmaxf(red[tid], red[tid + o]);
        __syncthreads();
    }
    mx = red[0]; __syncthreads();
    float sm = 0.f;
    for (int m = tid; m < NN; m += 256) {
        float ex = __expf(row[m] - mx);
        row[m] = ex; sm += ex;
    }
    red[tid] = sm; __syncthreads();
    for (int o = 128; o > 0; o >>= 1) {
        if (tid < o) red[tid] += red[tid + o];
        __syncthreads();
    }
    float inv = 1.f / red[0];
    for (int m = tid; m < NN; m += 256)
        Abf[(size_t)n * NN + m] = f2bf(row[m] * inv);
}

// ---------------------------------------------------------------------------
// Materialize per-node weights, transposed [n][o][kc], poly coef folded in.
// Wp layout: [16][3][C][O] f32.  kc = kk*CP + c ; c>=C rows are zero pad.
// ---------------------------------------------------------------------------
__global__ __launch_bounds__(256) void mat_w(const float* __restrict__ E,
                                             const float* __restrict__ Wp,
                                             const float* __restrict__ coef,
                                             __bf16* __restrict__ Wt,
                                             int C, int CP, int O) {
    const size_t KC = (size_t)3 * CP;
    size_t i = (size_t)blockIdx.x * 256 + threadIdx.x;
    size_t total = (size_t)NN * O * KC;
    if (i >= total) return;
    int kc = (int)(i % KC);
    size_t no = i / KC;
    int o = (int)(no % O);
    int n = (int)(no / O);
    int kk = kc / CP, c = kc % CP;
    float v = 0.f;
    if (c < C) {
#pragma unroll
        for (int d = 0; d < 16; d++)
            v += E[n * 16 + d] * Wp[(((size_t)d * 3 + kk) * C + c) * O + o];
        v *= coef[kk];
    }
    Wt[i] = f2bf(v);
}

__global__ __launch_bounds__(256) void mat_b(const float* __restrict__ E,
                                             const float* __restrict__ bp,
                                             float* __restrict__ bias, int O) {
    int i = blockIdx.x * 256 + threadIdx.x;
    if (i >= NN * O) return;
    int o = i % O, n = i / O;
    float v = 0.f;
#pragma unroll
    for (int d = 0; d < 16; d++) v += E[n * 16 + d] * bp[d * O + o];
    bias[i] = v;
}

__global__ __launch_bounds__(256) void init_h(const float* __restrict__ hs,
                                              float* __restrict__ h, int layer) {
    size_t i = (size_t)blockIdx.x * 256 + threadIdx.x;
    if (i >= (size_t)NN * BB * OO64) return;
    int o = (int)(i % OO64);
    size_t nb = i / OO64;
    int b = (int)(nb % BB), n = (int)(nb / BB);
    h[i] = hs[(((size_t)layer * BB + b) * NN + n) * OO64 + o];
}

// ---------------------------------------------------------------------------
// Pack xh = concat(xt, h, zeropad) directly into XG slice 0: [n][b][3*CP]
// ---------------------------------------------------------------------------
__global__ __launch_bounds__(256) void pack_xh(const float* __restrict__ xin,
                                               const __bf16* __restrict__ seq0,
                                               const float* __restrict__ h,
                                               __bf16* __restrict__ XG,
                                               int CP, int xtw, int t, int layer) {
    size_t i = (size_t)blockIdx.x * 256 + threadIdx.x;
    size_t total = (size_t)NN * BB * CP;
    if (i >= total) return;
    int c = (int)(i % CP);
    size_t nb = i / CP;
    int b = (int)(nb % BB), n = (int)(nb / BB);
    __bf16 v;
    if (c < xtw) {
        float xv = (layer == 0)
            ? xin[(((size_t)b * TT + t) * NN + n) * 2 + c]
            : bf2f(seq0[(((size_t)t * NN + n) * BB + b) * OO64 + c]);
        v = f2bf(xv);
    } else if (c < xtw + OO64) {
        v = f2bf(h[nb * OO64 + (c - xtw)]);
    } else {
        v = f2bf(0.f);
    }
    XG[nb * (size_t)(3 * CP) + c] = v;
}

// ---------------------------------------------------------------------------
// Graph GEMM:  out_slice = A(2048x2048 bf16) @ in_slice(2048 x 16*CP)
// in/out element (m, col=(b,c)):  ptr[(m*BB+b)*ld + off + c]   (slices of XG)
// Block tile 128x128, BK=32, 8 waves (4 along M x 2 along N), 2x4 WMMA/wave.
// Double-buffered LDS; B staged with vector global loads + packed b32 stores.
// ---------------------------------------------------------------------------
__global__ __launch_bounds__(256) void gemm_graph(const __bf16* __restrict__ A,
                                                  const __bf16* __restrict__ in,
                                                  __bf16* __restrict__ out,
                                                  int CP, int ldin, int in_off,
                                                  int ldout, int out_off) {
    __shared__ __align__(32) __bf16 As[2][128][48];   // [buf][m][k]
    __shared__ __align__(32) __bf16 Bs[2][128][48];   // [buf][col][k] (K-transposed)
    const int tid = threadIdx.x;
    const int lane = tid & 31, wave = tid >> 5;
    const int g = lane >> 4;
    const int wm = wave & 3, wn = wave >> 2;
    const int m0 = blockIdx.y * 128;
    const int n0 = blockIdx.x * 128;

    // --- A staging geometry: one 32B vector per thread half-row ---
    const int ar = tid >> 1;
    const int ac = (tid & 1) * 16;
    const __bf16* aptr = A + (size_t)(m0 + ar) * NN + ac;

    // --- B staging geometry: thread owns 8 consecutive cols x 2 k-rows ---
    // col groups of 8 never straddle a CP boundary (96,128 are multiples of 8)
    const int cg = (tid & 15) * 8;      // col group start (0..120)
    const int kp = (tid >> 4) * 2;      // k pair base (0..30)
    const int colg = n0 + cg;
    const int bb = colg / CP, cc = colg % CP;
    const __bf16* bptr = in + (size_t)bb * ldin + in_off + cc;
    const size_t brow = (size_t)BB * ldin;

    v8f acc[2][4];
#pragma unroll
    for (int i = 0; i < 2; i++)
#pragma unroll
        for (int j = 0; j < 4; j++)
            acc[i][j] = (v8f){0.f, 0.f, 0.f, 0.f, 0.f, 0.f, 0.f, 0.f};

    auto stage = [&](int k0, int buf) {
        // A tile: vector copy
        v16bf av = *(const v16bf*)(aptr + k0);
        *(v16bf*)(&As[buf][ar][ac]) = av;
        if (k0 + 64 < NN)
            __builtin_prefetch(aptr + k0 + 64, 0, 1);
        // B tile: two coalesced 16B row loads, 8 packed transposed b32 stores
        v8bf b0 = *(const v8bf*)(bptr + (size_t)(k0 + kp) * brow);
        v8bf b1 = *(const v8bf*)(bptr + (size_t)(k0 + kp + 1) * brow);
#pragma unroll
        for (int e = 0; e < 8; e++) {
            unsigned lo = __builtin_bit_cast(unsigned short, b0[e]);
            unsigned hi = __builtin_bit_cast(unsigned short, b1[e]);
            *(unsigned*)(&Bs[buf][cg + e][kp]) = lo | (hi << 16);
        }
    };

    stage(0, 0);
    __syncthreads();

    for (int it = 0; it < NN / 32; ++it) {
        const int cur = it & 1;
        if (it + 1 < NN / 32) stage((it + 1) * 32, cur ^ 1);

        v16bf af[2], bfr[4];
#pragma unroll
        for (int i = 0; i < 2; i++)
            af[i] = load_a_frag(&As[cur][wm * 32 + i * 16 + (lane & 15)][0], g);
#pragma unroll
        for (int j = 0; j < 4; j++)
            bfr[j] = load_b_frag(&Bs[cur][wn * 64 + j * 16 + (lane & 15)][0], g);
#pragma unroll
        for (int i = 0; i < 2; i++)
#pragma unroll
            for (int j = 0; j < 4; j++)
                acc[i][j] = wmma_bf16(af[i], bfr[j], acc[i][j]);
        __syncthreads();
    }

    // epilogue: f32 -> bf16 scatter into output slice
#pragma unroll
    for (int j = 0; j < 4; j++) {
        int col = n0 + wn * 64 + j * 16 + (lane & 15);
        int ob = col / CP, oc = col % CP;
#pragma unroll
        for (int i = 0; i < 2; i++) {
#pragma unroll
            for (int r = 0; r < 8; r++) {
                int m = m0 + wm * 32 + i * 16 + r + 8 * g;
                out[((size_t)m * BB + ob) * ldout + out_off + oc] = f2bf(acc[i][j][r]);
            }
        }
    }
}

// ---------------------------------------------------------------------------
// Per-node contraction: out[b,o] = Xg_n[16 x KC] @ W_n[KC x O] + bias
// GATE:   zr=sigmoid(.); o<64: write z*h (bf16) into XG slice0 mid region
//                        o>=64: write r (f32) to R
// UPDATE: hc=tanh(.); hn=r*h+(1-r)*hc; write h, seq0/d_out, hT at t==T-1
// ---------------------------------------------------------------------------
template <int O, bool GATE>
__global__ __launch_bounds__(256) void contract_kernel(
    const __bf16* __restrict__ xg, __bf16* __restrict__ xgw,
    const __bf16* __restrict__ Wt, const float* __restrict__ bias,
    float* __restrict__ h, float* __restrict__ R,
    __bf16* __restrict__ seq0, float* __restrict__ dout,
    int KC, int CP, int xtw, int t, int layer) {
    constexpr int NPB = 128 / O;   // nodes per block (1 or 2)
    constexpr int WPN = 8 / NPB;   // waves per node  (8 or 4) == O/16
    __shared__ __align__(32) __bf16 xs[2][16][400];

    const int tid = threadIdx.x, lane = tid & 31, wave = tid >> 5;
    const int g = lane >> 4;
    const int nodeBase = blockIdx.x * NPB;

    // stage this block's Xg rows ([16 x KC] per node) into LDS
    const int kc8 = KC / 8;
    for (int i = tid; i < NPB * 16 * kc8; i += 256) {
        int kv = i % kc8;
        int rb = i / kc8;
        int b = rb & 15, nl = rb >> 4;
        *(v8bf*)(&xs[nl][b][kv * 8]) =
            *(const v8bf*)(xg + ((size_t)(nodeBase + nl) * BB + b) * KC + kv * 8);
    }
    __syncthreads();

    const int nl = wave / WPN;
    const int ot = wave % WPN;
    const int n = nodeBase + nl;
    const int col_o = ot * 16 + (lane & 15);
    const __bf16* wcol = Wt + ((size_t)n * O + col_o) * KC;
    const __bf16* xrow = &xs[nl][lane & 15][0];

    v8f acc = (v8f){0.f, 0.f, 0.f, 0.f, 0.f, 0.f, 0.f, 0.f};
    for (int kc = 0; kc < KC; kc += 32) {
        v16bf a = load_a_frag(xrow + kc, g);
        v16bf b = load_b_frag(wcol + kc, g);
        acc = wmma_bf16(a, b, acc);
    }

    const float bv = bias[(size_t)n * O + col_o];
#pragma unroll
    for (int r = 0; r < 8; r++) {
        int b = r + 8 * g;                      // batch row
        size_t nb = (size_t)n * BB + b;
        float v = acc[r] + bv;
        if (GATE) {
            float s = 1.f / (1.f + __expf(-v));
            if (col_o < 64) {                   // z: write z*h into cand input
                float hv = h[nb * OO64 + col_o];
                xgw[nb * (size_t)(3 * CP) + xtw + col_o] = f2bf(s * hv);
            } else {                            // r gate
                R[nb * OO64 + (col_o - 64)] = s;
            }
        } else {
            float hc = tanhf(v);
            float rr = R[nb * OO64 + col_o];
            float hv = h[nb * OO64 + col_o];
            float hn = rr * hv + (1.f - rr) * hc;
            h[nb * OO64 + col_o] = hn;
            if (layer == 0)
                seq0[(((size_t)t * NN + n) * BB + b) * OO64 + col_o] = f2bf(hn);
            else
                dout[(((size_t)b * TT + t) * NN + n) * OO64 + col_o] = hn;
            if (t == TT - 1)
                dout[(size_t)BB * TT * NN * OO64 +
                     (((size_t)layer * BB + b) * NN + n) * OO64 + col_o] = hn;
        }
    }
}

// ---------------------------------------------------------------------------
extern "C" void kernel_launch(void* const* d_in, const int* in_sizes, int n_in,
                              void* d_out, int out_size, void* d_ws, size_t ws_size,
                              hipStream_t stream) {
    (void)in_sizes; (void)n_in; (void)out_size; (void)ws_size;
    const float* x    = (const float*)d_in[0];
    const float* hini = (const float*)d_in[1];
    const float* E    = (const float*)d_in[2];
    const float* coef = (const float*)d_in[3];
    const float* Wgp[2] = {(const float*)d_in[4], (const float*)d_in[8]};
    const float* bgp[2] = {(const float*)d_in[5], (const float*)d_in[9]};
    const float* Wup[2] = {(const float*)d_in[6], (const float*)d_in[10]};
    const float* bup[2] = {(const float*)d_in[7], (const float*)d_in[11]};
    float* out = (float*)d_out;

    char* w = (char*)d_ws;
    size_t off = 0;
    auto grab = [&](size_t bytes) -> char* {
        char* p = w + off;
        off = (off + bytes + 255) & ~(size_t)255;
        return p;
    };
    __bf16* Abf  = (__bf16*)grab((size_t)NN * NN * 2);            //   8 MB
    __bf16* XG   = (__bf16*)grab((size_t)NN * BB * 384 * 2);      //  25 MB
    __bf16* Wgt  = (__bf16*)grab((size_t)NN * 128 * 384 * 2);     // 201 MB
    __bf16* Wut  = (__bf16*)grab((size_t)NN * 64 * 384 * 2);      // 100 MB
    float*  bg   = (float*)grab((size_t)NN * 128 * 4);
    float*  bu   = (float*)grab((size_t)NN * 64 * 4);
    float*  h    = (float*)grab((size_t)NN * BB * OO64 * 4);
    float*  R    = (float*)grab((size_t)NN * BB * OO64 * 4);
    __bf16* seq0 = (__bf16*)grab((size_t)TT * NN * BB * OO64 * 2);// 100 MB

    adj_kernel<<<NN, 256, 0, stream>>>(E, Abf);

    for (int l = 0; l < 2; l++) {
        const int C   = l ? 128 : 66;
        const int CP  = l ? 128 : 96;    // padded channel count (3*CP % 32 == 0)
        const int KC  = 3 * CP;
        const int xtw = l ? 64 : 2;

        {   // per-layer weight materialization (poly coefs folded in)
            size_t tot = (size_t)NN * 128 * KC;
            mat_w<<<(unsigned)((tot + 255) / 256), 256, 0, stream>>>(E, Wgp[l], coef, Wgt, C, CP, 128);
            tot = (size_t)NN * 64 * KC;
            mat_w<<<(unsigned)((tot + 255) / 256), 256, 0, stream>>>(E, Wup[l], coef, Wut, C, CP, 64);
            mat_b<<<(NN * 128 + 255) / 256, 256, 0, stream>>>(E, bgp[l], bg, 128);
            mat_b<<<(NN * 64 + 255) / 256, 256, 0, stream>>>(E, bup[l], bu, 64);
            init_h<<<(NN * BB * OO64 + 255) / 256, 256, 0, stream>>>(hini, h, l);
        }

        const int cols = BB * CP;            // 1536 or 2048
        dim3 gg(cols / 128, NN / 128);
        const size_t ptot = (size_t)NN * BB * CP;
        const unsigned pblk = (unsigned)((ptot + 255) / 256);

        for (int t = 0; t < TT; t++) {
            // slice0 = concat(xt, h, pad)
            pack_xh<<<pblk, 256, 0, stream>>>(x, seq0, h, XG, CP, xtw, t, l);
            // slice1 = A @ slice0 ; slice2 = A @ slice1   (== A^2 @ slice0)
            gemm_graph<<<gg, 256, 0, stream>>>(Abf, XG, XG, CP, KC, 0,      KC, CP);
            gemm_graph<<<gg, 256, 0, stream>>>(Abf, XG, XG, CP, KC, CP,     KC, 2 * CP);
            // gate: zr = sigmoid(XG . Wg + bg); writes z*h into slice0, r -> R
            contract_kernel<128, true><<<NN, 256, 0, stream>>>(
                XG, XG, Wgt, bg, h, R, nullptr, nullptr, KC, CP, xtw, t, l);
            // candidate graph conv on concat(xt, z*h)
            gemm_graph<<<gg, 256, 0, stream>>>(Abf, XG, XG, CP, KC, 0,      KC, CP);
            gemm_graph<<<gg, 256, 0, stream>>>(Abf, XG, XG, CP, KC, CP,     KC, 2 * CP);
            // update: hc = tanh(XG . Wu + bu); h = r*h + (1-r)*hc; emit outputs
            contract_kernel<64, false><<<NN / 2, 256, 0, stream>>>(
                XG, XG, Wut, bu, h, R, seq0, out, KC, CP, xtw, t, l);
        }
    }
}
